// TimeSeriesStatisticalExtractor_76132590289184
// MI455X (gfx1250) — compile-verified, matrix operand order checked by reading, not run on hardware
//
#include <hip/hip_runtime.h>
#include <hip/hip_bf16.h>
#include <cstdint>
#include <cstddef>

#define EPSF  1e-8f
#define LNEPS 1e-5f

constexpr int Bx   = 512;
constexpr int T    = 2048;
constexpr int C    = 64;
constexpr int FD   = 256;
constexpr int STAT = C * 8 + C * (C - 1) / 2;   // 2528
constexpr int N1   = 2 * FD;                    // 512
constexpr int K1   = STAT;                      // 2528 = 79 * 32
constexpr int N2   = FD;                        // 256
constexpr int K2   = 2 * FD;                    // 512

typedef __attribute__((ext_vector_type(16))) __bf16 v16bf;
typedef __attribute__((ext_vector_type(8)))  float  v8f;
typedef __attribute__((ext_vector_type(4)))  unsigned int u32x4;
typedef __attribute__((ext_vector_type(8)))  unsigned int u32x8;

__device__ inline v8f wmma_bf16(v16bf a, v16bf b, v8f c) {
  // D = A(16x32 bf16) x B(32x16 bf16) + C(16x16 f32)
  return __builtin_amdgcn_wmma_f32_16x16x32_bf16(false, a, false, b, (short)0, c,
                                                 false, false);
}

// Build a v16bf operand from two 16-byte-aligned 16-byte chunks
// (K = base..base+7 and base+16..base+23 per lane-half, matching the
// 16-bit A/B WMMA VGPR layouts). Explicit <4 x i32> aligned loads ->
// encourage ds_load_b128 / global_load_b128.
__device__ inline v16bf load2x8(const __bf16* p0, const __bf16* p1) {
  const u32x4 lo = *(const u32x4*)__builtin_assume_aligned(p0, 16);
  const u32x4 hi = *(const u32x4*)__builtin_assume_aligned(p1, 16);
  const u32x8 w  = __builtin_shufflevector(lo, hi, 0, 1, 2, 3, 4, 5, 6, 7);
  v16bf r;
  __builtin_memcpy(&r, &w, 32);
  return r;
}

__device__ inline int   f2ord(float f) { int i = __float_as_int(f); return (i >= 0) ? i : (i ^ 0x7fffffff); }
__device__ inline float ord2f(int i)   { return __int_as_float((i >= 0) ? i : (i ^ 0x7fffffff)); }

// ------------------------------------------------------------------
// Kernel 1: one block per batch. Single streaming pass over x[b]:
//  - fp32 power sums S1..S4 + min/max per channel (register accum)
//  - bf16 transposed copy [c][t] into LDS per T-chunk
//  - 16 waves each own one 16x16 tile of the 64x64 Gram via WMMA bf16
//  - epilogue: moments -> feats, Gram -> upper-triangle correlations
// ------------------------------------------------------------------
constexpr int TC   = 256;       // T-chunk in LDS
constexpr int ROWP = TC + 8;    // +16B pad: 528B row stride (16B aligned, bank skew 4)

struct Smem1 {
  __bf16 xt[C][ROWP];           // 33792 B
  float  s1[C], s2[C], s3[C], s4[C];
  int    mni[C], mxi[C];
  float  mean[C], invs[C];
};

__global__ __launch_bounds__(512, 1)
void stats_gram_kernel(const float* __restrict__ x, float* __restrict__ feats) {
  __shared__ Smem1 sm;
  const int tid  = threadIdx.x;
  const int lane = tid & 31;
  const int wave = tid >> 5;            // 0..15
  const int b    = blockIdx.x;

  if (tid < C) {
    sm.s1[tid] = 0.f; sm.s2[tid] = 0.f; sm.s3[tid] = 0.f; sm.s4[tid] = 0.f;
    sm.mni[tid] = 0x7fffffff; sm.mxi[tid] = (int)0x80000000;
  }

  const int c0   = lane * 2;            // two channels per thread
  const int trow = wave;                // t sub-row 0..15
  float a1=0.f,a2=0.f,a3=0.f,a4=0.f, amn=3.4e38f, amx=-3.4e38f;
  float d1=0.f,d2=0.f,d3=0.f,d4=0.f, dmn=3.4e38f, dmx=-3.4e38f;

  v8f acc;
  #pragma unroll
  for (int i = 0; i < 8; ++i) acc[i] = 0.f;
  const int cA = (wave >> 2) << 4;      // Gram tile row block
  const int cB = (wave & 3)  << 4;      // Gram tile col block
  const int mr = lane & 15, kh = lane >> 4;

  const float* xb = x + (size_t)b * T * C;
  __syncthreads();

  for (int ch = 0; ch < T / TC; ++ch) {
    // coalesced load (float2 per thread), stats accum, bf16 transpose into LDS
    #pragma unroll 4
    for (int s = 0; s < TC / 16; ++s) {
      const int t = s * 16 + trow;
      const float2 v = *(const float2*)(xb + (size_t)(ch * TC + t) * C + c0);
      a1 += v.x; { const float xx = v.x * v.x; a2 += xx; a3 += xx * v.x; a4 += xx * xx; }
      amn = fminf(amn, v.x); amx = fmaxf(amx, v.x);
      d1 += v.y; { const float yy = v.y * v.y; d2 += yy; d3 += yy * v.y; d4 += yy * yy; }
      dmn = fminf(dmn, v.y); dmx = fmaxf(dmx, v.y);
      sm.xt[c0][t]     = (__bf16)v.x;
      sm.xt[c0 + 1][t] = (__bf16)v.y;
    }
    __syncthreads();
    // WMMA Gram over this chunk; A and B operands use identical per-lane addressing
    const __bf16* pa = &sm.xt[cA + mr][kh * 8];
    const __bf16* pb = &sm.xt[cB + mr][kh * 8];
    #pragma unroll
    for (int k0 = 0; k0 < TC; k0 += 32) {
      const v16bf av = load2x8(pa + k0, pa + k0 + 16);
      const v16bf bv = load2x8(pb + k0, pb + k0 + 16);
      acc = wmma_bf16(av, bv, acc);
    }
    __syncthreads();
  }

  // cross-wave stat reduction
  atomicAdd(&sm.s1[c0], a1); atomicAdd(&sm.s2[c0], a2);
  atomicAdd(&sm.s3[c0], a3); atomicAdd(&sm.s4[c0], a4);
  atomicMin(&sm.mni[c0], f2ord(amn)); atomicMax(&sm.mxi[c0], f2ord(amx));
  atomicAdd(&sm.s1[c0 + 1], d1); atomicAdd(&sm.s2[c0 + 1], d2);
  atomicAdd(&sm.s3[c0 + 1], d3); atomicAdd(&sm.s4[c0 + 1], d4);
  atomicMin(&sm.mni[c0 + 1], f2ord(dmn)); atomicMax(&sm.mxi[c0 + 1], f2ord(dmx));
  __syncthreads();

  float* fb = feats + (size_t)b * STAT;
  if (tid < C) {
    const int c = tid;
    const float S1 = sm.s1[c], S2 = sm.s2[c], S3 = sm.s3[c], S4 = sm.s4[c];
    const float Tf = (float)T;
    const float mean = S1 / Tf;
    const float css  = fmaxf(S2 - Tf * mean * mean, 0.f);          // sum centered^2
    const float var  = css / (Tf - 1.f);                           // ddof=1
    const float sd   = sqrtf(var);
    const float m3 = (S3 - 3.f * mean * S2 + 2.f * Tf * mean * mean * mean) / Tf;
    const float m4 = (S4 - 4.f * mean * S3 + 6.f * mean * mean * S2
                      - 3.f * Tf * mean * mean * mean * mean) / Tf;
    const float skew = m3 / (sd * sd * sd + EPSF);
    const float kurt = m4 / (var * var + EPSF);                    // std^4 == var^2
    const float cv   = sd / (fabsf(mean) + EPSF);
    fb[c]         = mean;
    fb[C + c]     = sd;
    fb[2 * C + c] = ord2f(sm.mxi[c]);
    fb[3 * C + c] = ord2f(sm.mni[c]);
    fb[5 * C + c] = skew;
    fb[6 * C + c] = kurt;
    fb[7 * C + c] = cv;
    sm.mean[c] = mean;
    sm.invs[c] = 1.f / (sd + EPSF);
  }
  __syncthreads();

  // Gram -> correlation, write triu(k=1) entries
  {
    const int nj = lane & 15;
    float* fc = fb + 8 * C;
    #pragma unroll
    for (int i = 0; i < 8; ++i) {
      const int ci = cA + i + 8 * kh;   // output row (VGPR layout)
      const int cj = cB + nj;           // output col
      if (ci < cj) {
        const float g = acc[i];
        const float corr = (g - (float)T * sm.mean[ci] * sm.mean[cj])
                           * sm.invs[ci] * sm.invs[cj] * (1.f / ((float)T - 1.f));
        const int idx = ci * (C - 1) - (ci * (ci - 1)) / 2 + (cj - ci - 1);
        fc[idx] = corr;
      }
    }
  }
}

// ------------------------------------------------------------------
// Kernel 2: exact median. One block per (b,c).
// The strided column (2048 fp32, stride 256B) is gathered into LDS by the
// Tensor Data Mover: 2D descriptor, tile_dim0=1 x tile_dim1=2048, walking
// tensor_dim0_stride=64 elements; TDM packs the tile contiguously at
// D#.lds_addr (=0, `arr` is the only LDS object). Then bitonic sort in LDS.
// ------------------------------------------------------------------
__global__ __launch_bounds__(256)
void median_kernel(const float* __restrict__ x, float* __restrict__ feats) {
  __shared__ float arr[T];              // only LDS object -> static LDS offset 0
  const int b = blockIdx.x >> 6;
  const int c = blockIdx.x & (C - 1);

  if (threadIdx.x < 32) {               // wave 0 issues the DMA (EXEC ignored by TDM)
    const unsigned long long ga =
        (unsigned long long)(const void*)(x + (size_t)b * T * C + c);
    u32x4 g0;
    g0[0] = 1u;                                        // count=1 user descriptor
    g0[1] = 0u;                                        // lds_addr (bytes)
    g0[2] = (unsigned int)(ga & 0xffffffffu);          // global_addr[31:0]
    g0[3] = (unsigned int)((ga >> 32) & 0x01ffffffu)   // global_addr[56:32]
            | (2u << 30);                              // D#.type = 2 (image)
    u32x8 g1;
    g1[0] = (2u << 16);                 // workgroup_mask=0, data_size=2 (4 bytes)
    g1[1] = ((unsigned int)C) << 16;    // tensor_dim0[15:0] = 64
    g1[2] = ((unsigned int)T) << 16;    // tensor_dim0[31:16]=0 | tensor_dim1[15:0]=2048
    g1[3] = (1u << 16);                 // tensor_dim1[31:16]=0 | tile_dim0 = 1
    g1[4] = (unsigned int)T;            // tile_dim1 = 2048 | tile_dim2 = 0
    g1[5] = (unsigned int)C;            // tensor_dim0_stride[31:0] = 64
    g1[6] = 0u;                         // stride0[47:32]=0 | tensor_dim1_stride lo=0
    g1[7] = 0u;
    // 2-group form (VADDR2/VADDR3 omitted): tensors up to 2D.
    asm volatile("tensor_load_to_lds %0, %1" :: "s"(g0), "s"(g1) : "memory");
    __builtin_amdgcn_s_wait_tensorcnt(0);   // data resident before the barrier
  }
  __syncthreads();

  for (int k = 2; k <= T; k <<= 1) {
    for (int j = k >> 1; j > 0; j >>= 1) {
      __syncthreads();
      for (int i = threadIdx.x; i < T; i += 256) {
        const int l = i ^ j;
        if (l > i) {
          const float u = arr[i], w = arr[l];
          const bool up = ((i & k) == 0);
          if (up ? (u > w) : (u < w)) { arr[i] = w; arr[l] = u; }
        }
      }
    }
  }
  __syncthreads();
  if (threadIdx.x == 0) feats[(size_t)b * STAT + 4 * C + c] = arr[(T - 1) / 2];
}

// ------------------------------------------------------------------
// fp32 -> bf16 conversion
// ------------------------------------------------------------------
__global__ __launch_bounds__(256)
void cvt_bf16_kernel(const float* __restrict__ in, __bf16* __restrict__ out, int n) {
  const int i = blockIdx.x * 256 + threadIdx.x;
  if (i < n) out[i] = (__bf16)in[i];
}

// ------------------------------------------------------------------
// GEMM1: h1 = relu(feats @ W1^T + b1), bf16 in / bf16 out, WMMA per 16x16 tile.
// A (M x K) row-major; Bw (N x K) row-major acts as B^T -> ideal B layout.
// ------------------------------------------------------------------
__global__ __launch_bounds__(256)
void gemm1_kernel(const __bf16* __restrict__ A, const __bf16* __restrict__ Bw,
                  const float* __restrict__ bias, __bf16* __restrict__ out) {
  const int lane = threadIdx.x & 31, wave = threadIdx.x >> 5;
  const int tile = blockIdx.x * 8 + wave;            // 32x32 = 1024 tiles
  const int tm = (tile >> 5) << 4;
  const int tn = (tile & 31) << 4;
  const int mr = lane & 15, kh = lane >> 4;
  const __bf16* pa = A  + (size_t)(tm + mr) * K1 + kh * 8;
  const __bf16* pb = Bw + (size_t)(tn + mr) * K1 + kh * 8;
  v8f acc;
  #pragma unroll
  for (int i = 0; i < 8; ++i) acc[i] = 0.f;
  for (int k0 = 0; k0 < K1; k0 += 32) {              // 79 steps
    const v16bf av = load2x8(pa + k0, pa + k0 + 16);
    const v16bf bv = load2x8(pb + k0, pb + k0 + 16);
    acc = wmma_bf16(av, bv, acc);
  }
  const int n = tn + mr;
  const float bv_ = bias[n];
  #pragma unroll
  for (int i = 0; i < 8; ++i) {
    const int m = tm + i + 8 * kh;
    out[(size_t)m * N1 + n] = (__bf16)fmaxf(acc[i] + bv_, 0.f);
  }
}

// ------------------------------------------------------------------
// GEMM2: h2 = h1 @ W2^T + b2, bf16 in / fp32 out.
// ------------------------------------------------------------------
__global__ __launch_bounds__(256)
void gemm2_kernel(const __bf16* __restrict__ A, const __bf16* __restrict__ Bw,
                  const float* __restrict__ bias, float* __restrict__ out) {
  const int lane = threadIdx.x & 31, wave = threadIdx.x >> 5;
  const int tile = blockIdx.x * 8 + wave;            // 32x16 = 512 tiles
  const int tm = (tile >> 4) << 4;
  const int tn = (tile & 15) << 4;
  const int mr = lane & 15, kh = lane >> 4;
  const __bf16* pa = A  + (size_t)(tm + mr) * K2 + kh * 8;
  const __bf16* pb = Bw + (size_t)(tn + mr) * K2 + kh * 8;
  v8f acc;
  #pragma unroll
  for (int i = 0; i < 8; ++i) acc[i] = 0.f;
  #pragma unroll 4
  for (int k0 = 0; k0 < K2; k0 += 32) {              // 16 steps
    const v16bf av = load2x8(pa + k0, pa + k0 + 16);
    const v16bf bv = load2x8(pb + k0, pb + k0 + 16);
    acc = wmma_bf16(av, bv, acc);
  }
  const int n = tn + mr;
  const float bv_ = bias[n];
  #pragma unroll
  for (int i = 0; i < 8; ++i) {
    const int m = tm + i + 8 * kh;
    out[(size_t)m * N2 + n] = acc[i] + bv_;
  }
}

// ------------------------------------------------------------------
// LayerNorm over last dim (FD=256): one block per row.
// ------------------------------------------------------------------
__global__ __launch_bounds__(256)
void ln_kernel(const float* __restrict__ h, const float* __restrict__ gamma,
               const float* __restrict__ beta, float* __restrict__ out) {
  const int b = blockIdx.x, i = threadIdx.x;
  const float v = h[(size_t)b * FD + i];
  float s = v, q = v * v;
  #pragma unroll
  for (int m = 16; m >= 1; m >>= 1) { s += __shfl_xor(s, m, 32); q += __shfl_xor(q, m, 32); }
  __shared__ float rs[8], rq[8];
  const int w = i >> 5, l = i & 31;
  if (l == 0) { rs[w] = s; rq[w] = q; }
  __syncthreads();
  if (w == 0) {
    float ss = (l < 8) ? rs[l] : 0.f;
    float qq = (l < 8) ? rq[l] : 0.f;
    #pragma unroll
    for (int m = 4; m >= 1; m >>= 1) { ss += __shfl_xor(ss, m, 32); qq += __shfl_xor(qq, m, 32); }
    if (l == 0) { rs[0] = ss; rq[0] = qq; }
  }
  __syncthreads();
  const float mu  = rs[0] * (1.f / FD);
  const float var = rq[0] * (1.f / FD) - mu * mu;
  out[(size_t)b * FD + i] = (v - mu) * rsqrtf(var + LNEPS) * gamma[i] + beta[i];
}

// ------------------------------------------------------------------
extern "C" void kernel_launch(void* const* d_in, const int* in_sizes, int n_in,
                              void* d_out, int out_size, void* d_ws, size_t ws_size,
                              hipStream_t stream) {
  const float* x     = (const float*)d_in[0];
  const float* W1    = (const float*)d_in[1];
  const float* b1    = (const float*)d_in[2];
  const float* W2    = (const float*)d_in[3];
  const float* b2    = (const float*)d_in[4];
  const float* gamma = (const float*)d_in[5];
  const float* beta  = (const float*)d_in[6];
  float* out = (float*)d_out;

  char* w = (char*)d_ws;
  float*  feats  = (float*)w;  w += (size_t)Bx * STAT * sizeof(float);
  __bf16* featsb = (__bf16*)w; w += (size_t)Bx * STAT * sizeof(__bf16);
  __bf16* W1b    = (__bf16*)w; w += (size_t)N1 * K1 * sizeof(__bf16);
  __bf16* W2b    = (__bf16*)w; w += (size_t)N2 * K2 * sizeof(__bf16);
  __bf16* h1b    = (__bf16*)w; w += (size_t)Bx * N1 * sizeof(__bf16);
  float*  h2     = (float*)w;  w += (size_t)Bx * N2 * sizeof(float);
  (void)ws_size; (void)in_sizes; (void)n_in; (void)out_size;

  stats_gram_kernel<<<Bx, 512, 0, stream>>>(x, feats);
  median_kernel<<<Bx * C, 256, 0, stream>>>(x, feats);
  cvt_bf16_kernel<<<(N1 * K1 + 255) / 256, 256, 0, stream>>>(W1, W1b, N1 * K1);
  cvt_bf16_kernel<<<(N2 * K2 + 255) / 256, 256, 0, stream>>>(W2, W2b, N2 * K2);
  cvt_bf16_kernel<<<(Bx * STAT + 255) / 256, 256, 0, stream>>>(feats, featsb, Bx * STAT);
  gemm1_kernel<<<(Bx / 16) * (N1 / 16) / 8, 256, 0, stream>>>(featsb, W1b, b1, h1b);
  gemm2_kernel<<<(Bx / 16) * (N2 / 16) / 8, 256, 0, stream>>>(h1b, W2b, b2, h2);
  ln_kernel<<<Bx, FD, 0, stream>>>(h2, gamma, beta, out);
}
